// LSTM_Decoder_51410758533280
// MI455X (gfx1250) — matrix operand order repaired and checked by reading, not compiled, not run on hardware
//
#include <hip/hip_runtime.h>
#include <hip/hip_bf16.h>

typedef __attribute__((ext_vector_type(16))) _Float16 v16h;
typedef __attribute__((ext_vector_type(8)))  _Float16 v8h;
typedef __attribute__((ext_vector_type(8)))  float    v8f;
typedef __attribute__((ext_vector_type(4)))  unsigned int u32x4;
typedef __attribute__((ext_vector_type(8)))  int i32x8;
typedef __attribute__((ext_vector_type(4)))  int i32x4;

#define LH    512      // hidden
#define NB    32       // batch
#define STEPS 81       // T-1
#define VV    32000
#define EE    256
#define NZV   64
#define G4H   2048     // 4*H
#define KIN   768      // E+H
#define RBLOCKS 16

__device__ __forceinline__ float sigm(float x){ return 1.0f/(1.0f+__expf(-x)); }

// ---- WMMA fragment helpers (CDNA5 16x16x32 f16, wave32 layouts, ISA 7.12.2) ----
__device__ __forceinline__ v16h frag_a_lds(const _Float16* base, int stride, int k0) {
  int lane = threadIdx.x & 31;
  const _Float16* row = base + (lane & 15) * stride + k0 + (lane >> 4) * 16;
  v8h lo = *(const v8h*)(row);
  v8h hi = *(const v8h*)(row + 8);
  v16h r;
#pragma unroll
  for (int j = 0; j < 8; ++j) { r[j] = lo[j]; r[8 + j] = hi[j]; }
  return r;
}

// B fragment for C = A @ W^T with W[N][K] row-major: one 32B contiguous load/lane.
__device__ __forceinline__ v16h frag_b_glb(const _Float16* W, int ldk, int n0, int k0) {
  int lane = threadIdx.x & 31;
  return *(const v16h*)(W + (size_t)(n0 + (lane & 15)) * ldk + k0 + (lane >> 4) * 16);
}

// ---- Tensor Data Mover: 2D 32x512 f16 tile, global(stride 512) -> LDS(stride 520) ----
// pad_enable with interval=256 DWORDs (code 7) + amount=4 DWORDs (code 3) implements the
// 520-f16 padded LDS row stride in hardware. LDS offset = low 32 bits of flat address.
__device__ __forceinline__ void tdm_load_h(const _Float16* gsrc, const _Float16* ldsdst) {
  unsigned ldsoff = (unsigned)(uintptr_t)ldsdst;   // shared aperture: [31:0] = LDS byte offset
  unsigned long long ga = (unsigned long long)(uintptr_t)gsrc;
  u32x4 g0;
  g0[0] = 1u;                                      // count=1 (valid), user mode
  g0[1] = ldsoff;                                  // lds_addr
  g0[2] = (unsigned)ga;                            // global_addr[31:0]
  g0[3] = (unsigned)(ga >> 32) | (2u << 30);       // global_addr[56:32] | type=2
  i32x8 g1;
  g1[0] = (1 << 16) | (1 << 20) | (7 << 22) | (3 << 25); // data_size=2B, pad_en, 256dw interval, 4dw pad
  g1[1] = (int)(512u << 16);                       // tensor_dim0 = 512 (bits 79:48 low half)
  g1[2] = (32 << 16);                              // tensor_dim1 = 32 (bits 111:80 low half)
  g1[3] = (512 << 16);                             // tile_dim0 = 512 (bits 127:112)
  g1[4] = 32;                                      // tile_dim1 = 32, tile_dim2 = 0
  g1[5] = 512;                                     // tensor_dim0_stride = 512
  g1[6] = 0; g1[7] = 0;
  i32x4 z4 = (i32x4)0;
#if defined(__clang_major__) && __clang_major__ >= 23
  i32x8 z8 = (i32x8)0;
  __builtin_amdgcn_tensor_load_to_lds(g0, g1, z4, z4, z8, 0);
#else
  __builtin_amdgcn_tensor_load_to_lds(g0, g1, z4, z4, 0);
#endif
}

// ---- pre-GEMM: G0[2592][2048] = X[2592][768] @ W_ih0[2048][768]^T + b0 ----
// A slab (32x768, 48.5KB) staged ONCE -> barrier-free k-loop.
#define PSTR 776   // 768+8, 776*2 = 97*16 bytes (16B aligned rows)
__global__ void __launch_bounds__(256) wmma_gemm_pre(
    const _Float16* __restrict__ A, const _Float16* __restrict__ B,
    const float* __restrict__ bias, float* __restrict__ C, int M, int N)
{
  __shared__ _Float16 sA[32 * PSTR];
  const int tid = threadIdx.x, wave = tid >> 5, lane = tid & 31;
  const int m0 = blockIdx.y * 32, nblk = blockIdx.x * 256;
  for (int c = tid; c < 32 * 96; c += 256) {       // 8-f16 chunks, 96 per row
    int r = c / 96, pos = (c % 96) * 8;
    *(v8h*)(sA + r * PSTR + pos) = *(const v8h*)(A + (size_t)(m0 + r) * KIN + pos);
  }
  __syncthreads();
  v8f acc[2][2];
#pragma unroll
  for (int mt = 0; mt < 2; ++mt)
#pragma unroll
    for (int nt = 0; nt < 2; ++nt) {
      float bv = bias[nblk + (wave * 2 + nt) * 16 + (lane & 15)];
#pragma unroll
      for (int r = 0; r < 8; ++r) acc[mt][nt][r] = bv;
    }
  for (int k0 = 0; k0 < KIN; k0 += 32) {
    v16h a0 = frag_a_lds(sA, PSTR, k0);
    v16h a1 = frag_a_lds(sA + 16 * PSTR, PSTR, k0);
    v16h b0 = frag_b_glb(B, KIN, nblk + (wave * 2 + 0) * 16, k0);
    v16h b1 = frag_b_glb(B, KIN, nblk + (wave * 2 + 1) * 16, k0);
    acc[0][0] = __builtin_amdgcn_wmma_f32_16x16x32_f16(false, a0, false, b0, (short)0, acc[0][0], false, false);
    acc[0][1] = __builtin_amdgcn_wmma_f32_16x16x32_f16(false, a0, false, b1, (short)0, acc[0][1], false, false);
    acc[1][0] = __builtin_amdgcn_wmma_f32_16x16x32_f16(false, a1, false, b0, (short)0, acc[1][0], false, false);
    acc[1][1] = __builtin_amdgcn_wmma_f32_16x16x32_f16(false, a1, false, b1, (short)0, acc[1][1], false, false);
  }
#pragma unroll
  for (int mt = 0; mt < 2; ++mt)
#pragma unroll
    for (int nt = 0; nt < 2; ++nt) {
      int n  = nblk + (wave * 2 + nt) * 16 + (lane & 15);
      int mb = m0 + mt * 16 + (lane >> 4) * 8;
#pragma unroll
      for (int r = 0; r < 8; ++r)
        C[(size_t)(mb + r) * N + n] = acc[mt][nt][r];
    }
}

// ---- output GEMM: logits[2592][32000] = outs[2592][512] @ W_out^T + b_out ----
// 96x256 block tile (2592 = 27*96 exact): 3x less L2 re-read of W_out vs 32-row tiles.
// A slab 96x512 (99.8KB LDS) staged once; 512 thr = 16 waves, wave w -> N-tile w, 6 M-tiles.
#define OSTR 520   // 512+8, rows 16B aligned, 4-bank row skew
__global__ void __launch_bounds__(512) wmma_gemm_out(
    const _Float16* __restrict__ A, const _Float16* __restrict__ B,
    const float* __restrict__ bias, float* __restrict__ C, int M, int N)
{
  __shared__ _Float16 sA[96 * OSTR];
  const int tid = threadIdx.x, wave = tid >> 5, lane = tid & 31;
  const int m0 = blockIdx.y * 96;
  const int n0 = blockIdx.x * 256 + wave * 16;
  for (int c = tid; c < 96 * 64; c += 512) {       // 8-f16 chunks, 64 per row
    int r = c >> 6, pos = (c & 63) * 8;
    *(v8h*)(sA + r * OSTR + pos) = *(const v8h*)(A + (size_t)(m0 + r) * LH + pos);
  }
  __syncthreads();
  v8f acc[6];
  {
    float bv = bias[n0 + (lane & 15)];
#pragma unroll
    for (int mt = 0; mt < 6; ++mt)
#pragma unroll
      for (int r = 0; r < 8; ++r) acc[mt][r] = bv;
  }
  for (int ks = 0; ks < 16; ++ks) {
    v16h b = frag_b_glb(B, LH, n0, ks * 32);
#pragma unroll
    for (int mt = 0; mt < 6; ++mt) {
      v16h a = frag_a_lds(sA + mt * 16 * OSTR, OSTR, ks * 32);
      acc[mt] = __builtin_amdgcn_wmma_f32_16x16x32_f16(false, a, false, b, (short)0, acc[mt], false, false);
    }
  }
#pragma unroll
  for (int mt = 0; mt < 6; ++mt) {
    int n  = n0 + (lane & 15);
    int mb = m0 + mt * 16 + (lane >> 4) * 8;
#pragma unroll
    for (int r = 0; r < 8; ++r)
      C[(size_t)(mb + r) * N + n] = acc[mt][r];
  }
}

// ---- prep: initial states (reshape scramble), fused biases, barrier init ----
__global__ void __launch_bounds__(256) prep_kernel(
    const float* __restrict__ z, const float* __restrict__ W_in, const float* __restrict__ b_in,
    const float* __restrict__ b_ih0, const float* __restrict__ b_hh0,
    const float* __restrict__ b_ih1, const float* __restrict__ b_hh1,
    float* __restrict__ cst, _Float16* __restrict__ h0h, _Float16* __restrict__ h1h,
    _Float16* __restrict__ zfh, float* __restrict__ b0, float* __restrict__ b1,
    int* __restrict__ bar)
{
  int tid = blockIdx.x * blockDim.x + threadIdx.x;
  if (tid < 2 * NB * LH) {
    int l = tid >> 14, b = (tid >> 9) & 31, h = tid & 511;
    int row = l * 16 + (b >> 1);
    int j   = (b & 1) * 512 + h;
    float v = b_in[j];
    const float* zr = z + row * NZV;
    const float* wr = W_in + (size_t)j * NZV;
    for (int k = 0; k < NZV; ++k) v += zr[k] * wr[k];
    cst[tid] = v;
    _Float16 th = (_Float16)tanhf(v);
    if (l == 0) h0h[b * LH + h] = th;
    else { h1h[b * LH + h] = th; zfh[b * LH + h] = (_Float16)v; }
  }
  if (tid < G4H) { b0[tid] = b_ih0[tid] + b_hh0[tid]; b1[tid] = b_ih1[tid] + b_hh1[tid]; }
  if (tid == 0) { bar[0] = 0; bar[1] = 0; }
}

__global__ void __launch_bounds__(256) cvt_f32_f16(const float* __restrict__ s,
                                                   _Float16* __restrict__ d, int n) {
  int i = blockIdx.x * blockDim.x + threadIdx.x;
  if (i < n) d[i] = (_Float16)s[i];
}

__global__ void __launch_bounds__(256) build_x(const int* __restrict__ text,
                                               const float* __restrict__ emb,
                                               const _Float16* __restrict__ zfh,
                                               _Float16* __restrict__ X) {
  int i = blockIdx.x * blockDim.x + threadIdx.x;
  if (i >= STEPS * NB * KIN) return;
  int c = i % KIN, row = i / KIN;
  int t = row >> 5, b = row & 31;
  if (c < EE) {
    int tok = (t == 0) ? 1 : text[t * NB + b];
    X[i] = (_Float16)emb[(size_t)tok * EE + c];
  } else {
    X[i] = zfh[b * LH + (c - EE)];
  }
}

// ---- grid-wide barrier (generation counter in workspace) ----
__device__ __forceinline__ void grid_sync(int* bar, int nblk) {
  __threadfence();
  __syncthreads();
  if (threadIdx.x == 0) {
    int g = __hip_atomic_load(bar + 1, __ATOMIC_ACQUIRE, __HIP_MEMORY_SCOPE_AGENT);
    if (__hip_atomic_fetch_add(bar, 1, __ATOMIC_ACQ_REL, __HIP_MEMORY_SCOPE_AGENT) == nblk - 1) {
      __hip_atomic_store(bar, 0, __ATOMIC_RELAXED, __HIP_MEMORY_SCOPE_AGENT);
      __hip_atomic_fetch_add(bar + 1, 1, __ATOMIC_RELEASE, __HIP_MEMORY_SCOPE_AGENT);
    } else {
      while (__hip_atomic_load(bar + 1, __ATOMIC_ACQUIRE, __HIP_MEMORY_SCOPE_AGENT) == g)
        __builtin_amdgcn_s_sleep(2);
    }
  }
  __syncthreads();
}

// ---- persistent recurrent kernel: 16 blocks x 512 thr = 256 waves (1 gate tile each) ----
#define SHS 520
__global__ void __launch_bounds__(512) recurrent_kernel(
    const float* __restrict__ G0,      // [2592][2048] precomputed x@W_ih0^T + b0
    const _Float16* __restrict__ Wh0,  // [2048][512]
    const _Float16* __restrict__ Wi1,  // [2048][512]
    const _Float16* __restrict__ Wh1,  // [2048][512]
    const float* __restrict__ b1,      // [2048]
    float* __restrict__ cst,           // [2][32][512]
    _Float16* __restrict__ h0h, _Float16* __restrict__ h1h,
    _Float16* __restrict__ outs,       // [2592][512]
    float* __restrict__ gbuf,          // [32][2048]
    int* __restrict__ bar)
{
  __shared__ _Float16 sH[2 * 32 * SHS];
  const int tid  = threadIdx.x;
  const int wave = tid >> 5, lane = tid & 31;
  const int tile = blockIdx.x * 16 + wave;   // 0..255
  const int mt   = tile & 1;
  const int n0   = (tile >> 1) * 16;
  const int nrow = n0 + (lane & 15);
  const int mrow = mt * 16 + (lane >> 4) * 8;

  for (int t = 0; t < STEPS; ++t) {
    // stage h0 -> LDS via Tensor Data Mover (hardware pad to 520-f16 stride)
    if (wave == 0) {
      tdm_load_h(h0h, sH);
      __builtin_amdgcn_s_wait_tensorcnt(0);
    }
    __syncthreads();

    // phase 1: gates0 = G0[t] + h0 @ Wh0^T   (M=32,N=2048,K=512)
    v8f acc;
#pragma unroll
    for (int r = 0; r < 8; ++r)
      acc[r] = G0[(size_t)(t * NB + mrow + r) * G4H + nrow];
    for (int ks = 0; ks < 16; ++ks) {
      v16h a = frag_a_lds(sH + mt * 16 * SHS, SHS, ks * 32);
      v16h b = frag_b_glb(Wh0, LH, n0, ks * 32);
      acc = __builtin_amdgcn_wmma_f32_16x16x32_f16(false, a, false, b, (short)0, acc, false, false);
    }
#pragma unroll
    for (int r = 0; r < 8; ++r)
      gbuf[(size_t)(mrow + r) * G4H + nrow] = acc[r];
    grid_sync(bar, RBLOCKS);

    // phase 2: layer-0 LSTM cell
    for (int e = blockIdx.x * 512 + tid; e < NB * LH; e += RBLOCKS * 512) {
      int b = e >> 9, h = e & 511;
      float gi = gbuf[b * G4H + h];
      float gf = gbuf[b * G4H + 512 + h];
      float gg = gbuf[b * G4H + 1024 + h];
      float go = gbuf[b * G4H + 1536 + h];
      float cc = sigm(gf) * cst[e] + sigm(gi) * tanhf(gg);
      cst[e] = cc;
      h0h[e] = (_Float16)(sigm(go) * tanhf(cc));
    }
    grid_sync(bar, RBLOCKS);

    // stage [h0n, h1] -> LDS via TDM
    if (wave == 0) {
      tdm_load_h(h0h, sH);
      tdm_load_h(h1h, sH + 32 * SHS);
      __builtin_amdgcn_s_wait_tensorcnt(0);
    }
    __syncthreads();

    // phase 3: gates1 = b1 + h0n @ Wi1^T + h1 @ Wh1^T  (K=1024 split)
#pragma unroll
    for (int r = 0; r < 8; ++r) acc[r] = b1[nrow];
    for (int ks = 0; ks < 32; ++ks) {
      int k0 = ks * 32, kk = k0 & 511;
      const _Float16* abase = sH + ((k0 < 512) ? 0 : 32 * SHS) + mt * 16 * SHS;
      const _Float16* W = (k0 < 512) ? Wi1 : Wh1;
      v16h a = frag_a_lds(abase, SHS, kk);
      v16h b = frag_b_glb(W, LH, n0, kk);
      acc = __builtin_amdgcn_wmma_f32_16x16x32_f16(false, a, false, b, (short)0, acc, false, false);
    }
#pragma unroll
    for (int r = 0; r < 8; ++r)
      gbuf[(size_t)(mrow + r) * G4H + nrow] = acc[r];
    grid_sync(bar, RBLOCKS);

    // phase 4: layer-1 LSTM cell; emit h1 (f16) for the output GEMM
    for (int e = blockIdx.x * 512 + tid; e < NB * LH; e += RBLOCKS * 512) {
      int b = e >> 9, h = e & 511;
      float gi = gbuf[b * G4H + h];
      float gf = gbuf[b * G4H + 512 + h];
      float gg = gbuf[b * G4H + 1024 + h];
      float go = gbuf[b * G4H + 1536 + h];
      float* c1 = cst + NB * LH;
      float cc = sigm(gf) * c1[e] + sigm(gi) * tanhf(gg);
      c1[e] = cc;
      _Float16 hh = (_Float16)(sigm(go) * tanhf(cc));
      h1h[e] = hh;
      outs[(size_t)(t * NB) * LH + e] = hh;
    }
    grid_sync(bar, RBLOCKS);
  }
}

extern "C" void kernel_launch(void* const* d_in, const int* in_sizes, int n_in,
                              void* d_out, int out_size, void* d_ws, size_t ws_size,
                              hipStream_t stream) {
  const float* z     = (const float*)d_in[0];
  const int*   text  = (const int*)d_in[1];
  const float* W_in  = (const float*)d_in[2];
  const float* b_in  = (const float*)d_in[3];
  const float* emb   = (const float*)d_in[4];
  const float* W_ih0 = (const float*)d_in[5];
  const float* W_hh0 = (const float*)d_in[6];
  const float* b_ih0 = (const float*)d_in[7];
  const float* b_hh0 = (const float*)d_in[8];
  const float* W_ih1 = (const float*)d_in[9];
  const float* W_hh1 = (const float*)d_in[10];
  const float* b_ih1 = (const float*)d_in[11];
  const float* b_hh1 = (const float*)d_in[12];
  const float* W_out = (const float*)d_in[13];
  const float* b_out = (const float*)d_in[14];
  float* out = (float*)d_out;

  char* cur = (char*)d_ws;
  auto alloc = [&](size_t bytes) -> void* {
    void* p = (void*)cur; cur += (bytes + 255) & ~(size_t)255; return p;
  };
  _Float16* Wih0h = (_Float16*)alloc((size_t)G4H * KIN * 2);
  _Float16* Whh0h = (_Float16*)alloc((size_t)G4H * LH * 2);
  _Float16* Wih1h = (_Float16*)alloc((size_t)G4H * LH * 2);
  _Float16* Whh1h = (_Float16*)alloc((size_t)G4H * LH * 2);
  _Float16* Wouth = (_Float16*)alloc((size_t)VV * LH * 2);
  _Float16* Xh    = (_Float16*)alloc((size_t)STEPS * NB * KIN * 2);
  float*    G0    = (float*)alloc((size_t)STEPS * NB * G4H * 4);
  _Float16* outsh = (_Float16*)alloc((size_t)STEPS * NB * LH * 2);
  float*    gbuf  = (float*)alloc((size_t)NB * G4H * 4);
  float*    b0    = (float*)alloc(G4H * 4);
  float*    b1    = (float*)alloc(G4H * 4);
  float*    cst   = (float*)alloc((size_t)2 * NB * LH * 4);
  _Float16* h0h   = (_Float16*)alloc((size_t)NB * LH * 2);
  _Float16* h1h   = (_Float16*)alloc((size_t)NB * LH * 2);
  _Float16* zfh   = (_Float16*)alloc((size_t)NB * LH * 2);
  int*      bar   = (int*)alloc(256);

  prep_kernel<<<128, 256, 0, stream>>>(z, W_in, b_in, b_ih0, b_hh0, b_ih1, b_hh1,
                                       cst, h0h, h1h, zfh, b0, b1, bar);
  auto cvt = [&](const float* s, _Float16* d, int n) {
    cvt_f32_f16<<<(n + 255) / 256, 256, 0, stream>>>(s, d, n);
  };
  cvt(W_ih0, Wih0h, G4H * KIN);
  cvt(W_hh0, Whh0h, G4H * LH);
  cvt(W_ih1, Wih1h, G4H * LH);
  cvt(W_hh1, Whh1h, G4H * LH);
  cvt(W_out, Wouth, VV * LH);
  {
    int n = STEPS * NB * KIN;
    build_x<<<(n + 255) / 256, 256, 0, stream>>>(text, emb, zfh, Xh);
  }
  wmma_gemm_pre<<<dim3(G4H / 256, (STEPS * NB) / 32), 256, 0, stream>>>(
      Xh, Wih0h, b0, G0, STEPS * NB, G4H);
  recurrent_kernel<<<RBLOCKS, 512, 0, stream>>>(G0, Whh0h, Wih1h, Whh1h, b1,
                                                cst, h0h, h1h, outsh, gbuf, bar);
  wmma_gemm_out<<<dim3(VV / 256, (STEPS * NB) / 96), 512, 0, stream>>>(
      outsh, Wouth, b_out, out, STEPS * NB, VV);
}
